// DSTP_RNN_52398601011721
// MI455X (gfx1250) — compile-verified, matrix operand order checked by reading, not run on hardware
//
#include <hip/hip_runtime.h>
#include <hip/hip_bf16.h>
#include <math.h>

// ---------------------------------------------------------------------------
// DSTP-RNN for MI455X (gfx1250, wave32, WMMA).
//  * Each block owns a 16-batch tile and runs the whole 128-step scan with
//    recurrent state + ALL weights resident in LDS (CDNA5: 320KB/WGP).
//  * Every per-step GEMM -> v_wmma_f32_16x16x32_f16 (f32 accumulate).
//  * GEMM operands live in LDS as packed f16, contiguous along K, so each
//    WMMA fragment is exactly two 16B ds_load_b128 per lane (ISA 7.12.2:
//    lane ks = {kb+half*8+[0..8), kb+16+half*8+[0..8)}). No per-step cvt.
//  * f32 cell state kept separately so the 128-step recurrence accumulates
//    in full precision; f16 copies only feed the WMMA (f16 boundary anyway).
//  * encoded/Uenc (34MB) live in workspace and stay L2-resident (192MB).
// ---------------------------------------------------------------------------

typedef __attribute__((ext_vector_type(16))) _Float16 v16h;
typedef __attribute__((ext_vector_type(8)))  _Float16 v8h;
typedef __attribute__((ext_vector_type(8)))  float    v8f;

#define DEV __device__ __forceinline__

// Problem constants
#define BATCH 512
#define TT    128   // T
#define NN    81    // N = 9*9
#define MM    64    // M
#define PP    64    // P
#define CC    8     // C
#define BT    16    // batch tile per block

DEV float sigm(float x) { return 1.0f / (1.0f + __expf(-x)); }

// Fragment load from K-contiguous f16 row (16B aligned): two b128 LDS loads.
DEV v16h load_frag16(const _Float16* row, int kbase, int half) {
  v8h lo = *(const v8h*)(row + kbase + half * 8);
  v8h hi = *(const v8h*)(row + kbase + 16 + half * 8);
  return __builtin_shufflevector(lo, hi, 0, 1, 2, 3, 4, 5, 6, 7, 8, 9, 10, 11,
                                 12, 13, 14, 15);
}

// Generic (gather) fragment loader, used only by the one-shot Uenc GEMM.
template <typename F>
DEV v16h load_frag(F f, int kbase, int half) {
  v16h x;
#pragma unroll
  for (int e = 0; e < 16; ++e) {
    int v = e >> 1;
    int k = kbase + ((v >> 2) << 4) + (half << 3) + ((v & 3) << 1) + (e & 1);
    x[e] = (_Float16)f(k);
  }
  return x;
}

DEV v8f wmma_f16(v16h a, v16h b, v8f c) {
  return __builtin_amdgcn_wmma_f32_16x16x32_f16(false, a, false, b, (short)0, c,
                                                false, false);
}

// ===========================================================================
// Encoder: conv features + input attention + LSTM, 128 sequential steps.
// grid = 32 blocks (16 batches each), block = 256 threads = 8 waves.
// ===========================================================================
#define ENC_SMEM_BYTES 240960
__global__ void __launch_bounds__(256) enc_kernel(
    const float* __restrict__ X, const float* __restrict__ Kw,
    const float* __restrict__ Kb, const float* __restrict__ We,
    const float* __restrict__ Web, const float* __restrict__ Ue,
    const float* __restrict__ ve, const float* __restrict__ Wih,
    const float* __restrict__ Whh, const float* __restrict__ bih,
    const float* __restrict__ bhh, float* __restrict__ encoded) {
  extern __shared__ char smc[];
  // ---- f16 region (all rows 16B aligned, K-contiguous) ----
  _Float16* lWe  = (_Float16*)smc;   // [128][128] We (f16)         16384
  _Float16* lGw  = lWe + 16384;      // [256][96]  [Wih|Whh|pad]    24576
  _Float16* lUeP = lGw + 24576;      // [128][32]  Ue padded K=32    4096
  _Float16* sCtH = lUeP + 4096;      // [1024][32] c_t padded K=32  32768
  _Float16* sHS  = sCtH + 32768;     // [16][128]  [h|s] mirror      2048
  _Float16* sA   = sHS + 2048;       // [16][96]   [w_in|h|pad]      1536
  // ---- f32 region ----
  float* lve  = (float*)(sA + 1536); // 128
  float* lWeb = lve + 128;           // 128
  float* lB   = lWeb + 128;          // 256 (bih+bhh)
  float* lKw  = lB + 256;            // 64x18 = 1152
  float* lKb  = lKw + 1152;          // 64
  float* sXt  = lKb + 64;            // 16x81 = 1296
  float* sCt  = sXt + 1296;          // 16x64x8 = 8192
  float* sC   = sCt + 8192;          // 16x64 cell state (f32) = 1024
  float* sX   = sC + 1024;           // 16x128 = 2048
  float* sE   = sX + 2048;           // 16x64 = 1024
  float* sWin = sE + 1024;           // 16x8 = 128
  float* sG   = sWin + 128;          // 16x256 = 4096

  const int tid  = threadIdx.x;
  const int wave = tid >> 5;
  const int lane = tid & 31;
  const int half = lane >> 4;
  const int lm   = lane & 15;
  const int bg0  = blockIdx.x * BT;

  __builtin_prefetch(We, 0, 3);
  __builtin_prefetch(Whh, 0, 3);

  // ---- stage weights into LDS once (reused 128x), downconvert to f16 ----
  for (int i = tid; i < 16384; i += 256) lWe[i] = (_Float16)We[i];
  for (int i = tid; i < 24576; i += 256) {
    int n = i / 96, k = i % 96;
    float v = k < 8 ? Wih[n * 8 + k] : (k < 72 ? Whh[n * 64 + (k - 8)] : 0.f);
    lGw[i] = (_Float16)v;
  }
  for (int i = tid; i < 4096; i += 256) {
    int n = i >> 5, k = i & 31;
    lUeP[i] = (_Float16)(k < 8 ? Ue[n * 8 + k] : 0.f);
  }
  for (int i = tid; i < 32768; i += 256) sCtH[i] = (_Float16)0.f;  // K padding
  for (int i = tid; i < 2048; i += 256) sHS[i] = (_Float16)0.f;    // h=s=0
  for (int i = tid; i < 1536; i += 256) sA[i] = (_Float16)0.f;
  for (int i = tid; i < 1024; i += 256) sC[i] = 0.f;
  if (tid < 128) { lve[tid] = ve[tid]; lWeb[tid] = Web[tid]; }
  lB[tid] = bih[tid] + bhh[tid];
  for (int i = tid; i < 1152; i += 256) lKw[i] = Kw[i];
  if (tid < 64) lKb[tid] = Kb[tid];
  __syncthreads();

  for (int t = 0; t < TT; ++t) {
    // ---- load X tile for this timestep ----
    for (int i = tid; i < BT * NN; i += 256) {
      int b = i / NN, n = i % NN;
      sXt[i] = X[(size_t)(bg0 + b) * (TT * NN) + (size_t)t * NN + n];
    }
    __syncthreads();

    // ---- conv features: relu -> sCt (f32) + sCtH (f16, K-padded) ----
    for (int i = tid; i < BT * MM * CC; i += 256) {
      int b = i >> 9, m = (i >> 3) & 63, r = i & 7;
      const float* xr = &sXt[b * NN + r * 9];
      const float* k0 = &lKw[m * 18];
      float v = lKb[m];
#pragma unroll
      for (int w = 0; w < 9; ++w) v += xr[w] * k0[w] + xr[9 + w] * k0[9 + w];
      v = v > 0.f ? v : 0.f;
      sCt[i] = v;
      sCtH[(i >> 3) * 32 + (i & 7)] = (_Float16)v;
    }

    // ---- x = [h|s](16x128) @ We^T + Web : 8 waves x 1 column tile ----
    {
      const _Float16* arow = &sHS[lm * 128];
      const int n = wave * 16 + lm;
      const _Float16* brow = &lWe[n * 128];
      v8f acc = {};
#pragma unroll
      for (int kb = 0; kb < 128; kb += 32)
        acc = wmma_f16(load_frag16(arow, kb, half), load_frag16(brow, kb, half),
                       acc);
#pragma unroll
      for (int r = 0; r < 8; ++r)
        sX[(r + 8 * half) * 128 + n] = acc[r] + lWeb[n];
    }
    __syncthreads();

    // ---- attention scores: e[b,m] = ve . tanh(x[b] + Ue c_t[b,m]) ----
    // rows = 1024 (b,m) pairs -> 64 row tiles; fuse tanh+dot in C fragment.
    for (int rt = wave; rt < 64; rt += 8) {
      const int rr = rt * 16 + lm;               // A row for this lane
      const _Float16* arow = &sCtH[rr * 32];     // padded to K=32
      float es[8];
#pragma unroll
      for (int r = 0; r < 8; ++r) es[r] = 0.f;
      for (int ct = 0; ct < 8; ++ct) {
        const int n = ct * 16 + lm;
        v8f acc = {};
        acc = wmma_f16(load_frag16(arow, 0, half),
                       load_frag16(&lUeP[n * 32], 0, half), acc);
        const float vev = lve[n];
#pragma unroll
        for (int r = 0; r < 8; ++r) {
          int grow = rt * 16 + r + 8 * half;     // global (b*64+m) row
          int b2 = grow >> 6;
          es[r] += tanhf(acc[r] + sX[b2 * 128 + n]) * vev;
        }
      }
      // reduce over the 16 lanes that share the same rows
#pragma unroll
      for (int r = 0; r < 8; ++r) {
#pragma unroll
        for (int off = 1; off < 16; off <<= 1)
          es[r] += __shfl_xor(es[r], off, 32);
      }
      if (lm == 0) {
#pragma unroll
        for (int r = 0; r < 8; ++r) sE[rt * 16 + r + 8 * half] = es[r];
      }
    }
    __syncthreads();

    // ---- softmax over m (64) per batch ----
    if (tid < BT) {
      float mx = -1e30f;
      for (int m = 0; m < 64; ++m) mx = fmaxf(mx, sE[tid * 64 + m]);
      float s = 0.f;
      for (int m = 0; m < 64; ++m) {
        float v = __expf(sE[tid * 64 + m] - mx);
        sE[tid * 64 + m] = v;
        s += v;
      }
      float inv = 1.f / s;
      for (int m = 0; m < 64; ++m) sE[tid * 64 + m] *= inv;
    }
    __syncthreads();

    // ---- w_in[b,c] = sum_m alpha[b,m] * c_t[b,m,c] (f32) -> sA f16 ----
    if (tid < BT * CC) {
      int b = tid >> 3, c = tid & 7;
      float s = 0.f;
      for (int m = 0; m < 64; ++m)
        s += sE[b * 64 + m] * sCt[(b * 64 + m) * 8 + c];
      sWin[tid] = s;
      sA[b * 96 + c] = (_Float16)s;   // LSTM A row: [w_in|h|pad]
    }
    __syncthreads();

    // ---- LSTM gates: g(16x256) = sA(16x96) @ lGw^T + b ----
    for (int nt = wave; nt < 16; nt += 8) {
      const int n = nt * 16 + lm;
      const _Float16* arow = &sA[lm * 96];
      const _Float16* brow = &lGw[n * 96];
      v8f acc = {};
#pragma unroll
      for (int kb = 0; kb < 96; kb += 32)
        acc = wmma_f16(load_frag16(arow, kb, half), load_frag16(brow, kb, half),
                       acc);
#pragma unroll
      for (int r = 0; r < 8; ++r)
        sG[(r + 8 * half) * 256 + n] = acc[r] + lB[n];
    }
    __syncthreads();

    // ---- cell update (f32 state) + emit h; refresh f16 mirrors ----
    for (int i = tid; i < BT * MM; i += 256) {
      int b = i >> 6, j = i & 63;
      float gi = sG[b * 256 + j];
      float gf = sG[b * 256 + 64 + j];
      float gg = sG[b * 256 + 128 + j];
      float go = sG[b * 256 + 192 + j];
      float c = sigm(gf) * sC[i] + sigm(gi) * tanhf(gg);
      float h = sigm(go) * tanhf(c);
      sC[i] = c;
      sHS[b * 128 + j] = (_Float16)h;
      sHS[b * 128 + 64 + j] = (_Float16)c;
      sA[b * 96 + 8 + j] = (_Float16)h;
      encoded[(size_t)(bg0 + b) * (TT * MM) + (size_t)t * MM + j] = h;
    }
    __syncthreads();
  }
}

// ===========================================================================
// Uenc = encoded @ Ud^T : (B*T x 64) @ (64 x 64). One-shot WMMA GEMM.
// grid = 4096 row tiles, block = 128 (4 waves = 4 column tiles).
// ===========================================================================
__global__ void __launch_bounds__(128) uenc_kernel(
    const float* __restrict__ enc, const float* __restrict__ Ud,
    float* __restrict__ Uenc) {
  const int wave = threadIdx.x >> 5;
  const int lane = threadIdx.x & 31;
  const int half = lane >> 4;
  const int lm   = lane & 15;
  const size_t rt = blockIdx.x;  // 0..4095

  const float* arow = enc + (rt * 16 + lm) * 64;
  const int n = wave * 16 + lm;
  const float* brow = Ud + n * 64;
  v8f acc = {};
#pragma unroll
  for (int kb = 0; kb < 64; kb += 32) {
    v16h a = load_frag([&](int k) { return arow[k]; }, kb, half);
    v16h b = load_frag([&](int k) { return brow[k]; }, kb, half);
    acc = wmma_f16(a, b, acc);
  }
#pragma unroll
  for (int r = 0; r < 8; ++r)
    Uenc[(rt * 16 + r + 8 * half) * 64 + n] = acc[r];
}

// ===========================================================================
// Decoder: temporal attention over Uenc/encoded + LSTM, 128 sequential steps.
// grid = 32 blocks (16 batches each), block = 256.
// ===========================================================================
#define DEC_SMEM_BYTES 111432
__global__ void __launch_bounds__(256) dec_kernel(
    const float* __restrict__ Y, const float* __restrict__ Wd,
    const float* __restrict__ Wdb, const float* __restrict__ vd,
    const float* __restrict__ wt, const float* __restrict__ wtb,
    const float* __restrict__ Wih, const float* __restrict__ Whh,
    const float* __restrict__ bih, const float* __restrict__ bhh,
    const float* __restrict__ encoded, const float* __restrict__ Uenc,
    float* __restrict__ c_last, float* __restrict__ d_final) {
  extern __shared__ char smc[];
  // ---- f16 region ----
  _Float16* lWd = (_Float16*)smc;  // [64][128] Wd (f16)            8192
  _Float16* lGw = lWd + 8192;      // [256][96] [Wih|Whh|pad]      24576
  _Float16* sDS = lGw + 24576;     // [16][128] [d|sp] mirror       2048
  _Float16* sA  = sDS + 2048;      // [16][96]  [y_tilda|d|pad]     1536
  // ---- f32 region ----
  float* lvd  = (float*)(sA + 1536);  // 64
  float* lwt  = lvd + 64;             // 66 (wt_w[65] + wt_b)
  float* lB   = lwt + 66;             // 256
  float* lWdb = lB + 256;             // 64
  float* sX1  = lWdb + 64;            // 16x64 = 1024
  float* sL   = sX1 + 1024;           // 16x128 = 2048
  float* sCt  = sL + 2048;            // 16x64 = 1024
  float* sYt  = sCt + 1024;           // 16
  float* sC   = sYt + 16;             // 16x64 cell state (f32)
  float* sG   = sC + 1024;            // 16x256 = 4096

  const int tid  = threadIdx.x;
  const int wave = tid >> 5;
  const int lane = tid & 31;
  const int half = lane >> 4;
  const int lm   = lane & 15;
  const int bg0  = blockIdx.x * BT;

  __builtin_prefetch(Whh, 0, 3);

  for (int i = tid; i < 8192; i += 256) lWd[i] = (_Float16)Wd[i];
  for (int i = tid; i < 24576; i += 256) {
    int n = i / 96, k = i % 96;
    float v = k == 0 ? Wih[n] : (k < 65 ? Whh[n * 64 + (k - 1)] : 0.f);
    lGw[i] = (_Float16)v;
  }
  for (int i = tid; i < 2048; i += 256) sDS[i] = (_Float16)0.f;  // d=sp=0
  for (int i = tid; i < 1536; i += 256) sA[i] = (_Float16)0.f;
  for (int i = tid; i < 1024; i += 256) sC[i] = 0.f;
  if (tid < 64) { lvd[tid] = vd[tid]; lWdb[tid] = Wdb[tid]; }
  if (tid < 65) lwt[tid] = wt[tid];
  if (tid == 0) lwt[65] = wtb[0];
  lB[tid] = bih[tid] + bhh[tid];
  __syncthreads();

  for (int t = 0; t < TT; ++t) {
    // ---- x1 = [d|sp](16x128) @ Wd^T + Wdb : waves 0..3, N=64 ----
    if (wave < 4) {
      const _Float16* arow = &sDS[lm * 128];
      const int n = wave * 16 + lm;
      const _Float16* brow = &lWd[n * 128];
      v8f acc = {};
#pragma unroll
      for (int kb = 0; kb < 128; kb += 32)
        acc = wmma_f16(load_frag16(arow, kb, half), load_frag16(brow, kb, half),
                       acc);
#pragma unroll
      for (int r = 0; r < 8; ++r)
        sX1[(r + 8 * half) * 64 + n] = acc[r] + lWdb[n];
    }
    __syncthreads();

    // ---- l[b,t'] = vd . tanh(x1[b] + Uenc[b,t']) ----
    for (int task = tid; task < BT * TT; task += 256) {
      int b = task >> 7, tp = task & 127;
      const float* up = &Uenc[((size_t)(bg0 + b) * TT + tp) * 64];
      const float* xp = &sX1[b * 64];
      float s = 0.f;
      for (int j = 0; j < 64; ++j) s += lvd[j] * tanhf(xp[j] + up[j]);
      sL[task] = s;
    }
    __syncthreads();

    // ---- softmax over t' (128) per batch ----
    if (tid < BT) {
      float mx = -1e30f;
      for (int tp = 0; tp < 128; ++tp) mx = fmaxf(mx, sL[tid * 128 + tp]);
      float s = 0.f;
      for (int tp = 0; tp < 128; ++tp) {
        float v = __expf(sL[tid * 128 + tp] - mx);
        sL[tid * 128 + tp] = v;
        s += v;
      }
      float inv = 1.f / s;
      for (int tp = 0; tp < 128; ++tp) sL[tid * 128 + tp] *= inv;
    }
    __syncthreads();

    // ---- c_t[b,j] = sum_t' beta[b,t'] * encoded[b,t',j] (L2-resident) ----
    for (int i = tid; i < BT * MM; i += 256) {
      int b = i >> 6, j = i & 63;
      const float* ep = &encoded[((size_t)(bg0 + b) * TT) * 64 + j];
      float s = 0.f;
      for (int tp = 0; tp < 128; ++tp) s += sL[b * 128 + tp] * ep[tp * 64];
      sCt[i] = s;
    }
    __syncthreads();

    // ---- y_tilda ----
    if (tid < BT) {
      float s = 0.f;
      for (int j = 0; j < 64; ++j) s += sCt[tid * 64 + j] * lwt[j];
      s += Y[(size_t)(bg0 + tid) * TT + t] * lwt[64] + lwt[65];
      sYt[tid] = s;
      sA[tid * 96] = (_Float16)s;   // LSTM A row: [y_tilda|d|pad]
    }
    __syncthreads();

    // ---- LSTM gates: g(16x256) = sA(16x96) @ lGw^T + b ----
    for (int nt = wave; nt < 16; nt += 8) {
      const int n = nt * 16 + lm;
      const _Float16* arow = &sA[lm * 96];
      const _Float16* brow = &lGw[n * 96];
      v8f acc = {};
#pragma unroll
      for (int kb = 0; kb < 96; kb += 32)
        acc = wmma_f16(load_frag16(arow, kb, half), load_frag16(brow, kb, half),
                       acc);
#pragma unroll
      for (int r = 0; r < 8; ++r)
        sG[(r + 8 * half) * 256 + n] = acc[r] + lB[n];
    }
    __syncthreads();

    // ---- cell update (f32 state) ----
    for (int i = tid; i < BT * PP; i += 256) {
      int b = i >> 6, j = i & 63;
      float gi = sG[b * 256 + j];
      float gf = sG[b * 256 + 64 + j];
      float gg = sG[b * 256 + 128 + j];
      float go = sG[b * 256 + 192 + j];
      float c = sigm(gf) * sC[i] + sigm(gi) * tanhf(gg);
      float d = sigm(go) * tanhf(c);
      sC[i] = c;
      sDS[b * 128 + j] = (_Float16)d;
      sDS[b * 128 + 64 + j] = (_Float16)c;
      sA[b * 96 + 1 + j] = (_Float16)d;
      if (t == TT - 1) {
        c_last[(size_t)(bg0 + b) * 64 + j] = sCt[b * 64 + j];
        d_final[(size_t)(bg0 + b) * 64 + j] = d;
      }
    }
    __syncthreads();
  }
}

// ===========================================================================
// Output head: out[b] = vy . (Wy @ [d|c_last] + Wyb) + vyb
// ===========================================================================
__global__ void __launch_bounds__(64) out_kernel(
    const float* __restrict__ dfin, const float* __restrict__ clast,
    const float* __restrict__ Wy, const float* __restrict__ Wyb,
    const float* __restrict__ vy, const float* __restrict__ vyb,
    float* __restrict__ out) {
  __shared__ float red[64];
  const int b = blockIdx.x, j = threadIdx.x;
  const float* wr = &Wy[j * 128];
  float s = 0.f;
  for (int k = 0; k < 64; ++k) s += dfin[(size_t)b * 64 + k] * wr[k];
  for (int k = 0; k < 64; ++k) s += clast[(size_t)b * 64 + k] * wr[64 + k];
  red[j] = (s + Wyb[j]) * vy[j];
  __syncthreads();
  if (j == 0) {
    float tacc = 0.f;
    for (int k = 0; k < 64; ++k) tacc += red[k];
    out[b] = tacc + vyb[0];
  }
}

// ===========================================================================
extern "C" void kernel_launch(void* const* d_in, const int* in_sizes, int n_in,
                              void* d_out, int out_size, void* d_ws,
                              size_t ws_size, hipStream_t stream) {
  (void)in_sizes; (void)n_in; (void)out_size; (void)ws_size;
  const float* X        = (const float*)d_in[0];
  const float* Y        = (const float*)d_in[1];
  const float* conv_w   = (const float*)d_in[2];
  const float* conv_b   = (const float*)d_in[3];
  const float* We_w     = (const float*)d_in[4];
  const float* We_b     = (const float*)d_in[5];
  const float* Ue_w     = (const float*)d_in[6];
  const float* ve_w     = (const float*)d_in[7];
  const float* enc_Wih  = (const float*)d_in[8];
  const float* enc_Whh  = (const float*)d_in[9];
  const float* enc_bih  = (const float*)d_in[10];
  const float* enc_bhh  = (const float*)d_in[11];
  const float* Wd_w     = (const float*)d_in[12];
  const float* Wd_b     = (const float*)d_in[13];
  const float* Ud_w     = (const float*)d_in[14];
  const float* vd_w     = (const float*)d_in[15];
  const float* wt_w     = (const float*)d_in[16];
  const float* wt_b     = (const float*)d_in[17];
  const float* dec_Wih  = (const float*)d_in[18];
  const float* dec_Whh  = (const float*)d_in[19];
  const float* dec_bih  = (const float*)d_in[20];
  const float* dec_bhh  = (const float*)d_in[21];
  const float* Wy_w     = (const float*)d_in[22];
  const float* Wy_b     = (const float*)d_in[23];
  const float* vy_w     = (const float*)d_in[24];
  const float* vy_b     = (const float*)d_in[25];

  // workspace layout (floats): encoded | Uenc | c_last | d_final  (~33.8 MB)
  float* ws       = (float*)d_ws;
  float* encoded  = ws;
  float* Uenc     = encoded + (size_t)BATCH * TT * MM;
  float* c_last   = Uenc + (size_t)BATCH * TT * MM;
  float* d_final  = c_last + (size_t)BATCH * MM;

  // CDNA5 allows up to 320KB LDS per workgroup; opt in past the 64KB default.
  (void)hipFuncSetAttribute((const void*)enc_kernel,
                            hipFuncAttributeMaxDynamicSharedMemorySize,
                            ENC_SMEM_BYTES);
  (void)hipFuncSetAttribute((const void*)dec_kernel,
                            hipFuncAttributeMaxDynamicSharedMemorySize,
                            DEC_SMEM_BYTES);

  enc_kernel<<<BATCH / BT, 256, ENC_SMEM_BYTES, stream>>>(
      X, conv_w, conv_b, We_w, We_b, Ue_w, ve_w, enc_Wih, enc_Whh, enc_bih,
      enc_bhh, encoded);
  uenc_kernel<<<(BATCH * TT) / 16, 128, 0, stream>>>(encoded, Ud_w, Uenc);
  dec_kernel<<<BATCH / BT, 256, DEC_SMEM_BYTES, stream>>>(
      Y, Wd_w, Wd_b, vd_w, wt_w, wt_b, dec_Wih, dec_Whh, dec_bih, dec_bhh,
      encoded, Uenc, c_last, d_final);
  out_kernel<<<BATCH, 64, 0, stream>>>(d_final, c_last, Wy_w, Wy_b, vy_w, vy_b,
                                       (float*)d_out);
}
